// ProductQuantizer_17540646437247
// MI455X (gfx1250) — matrix operand order repaired and checked by reading, not compiled, not run on hardware
//
#include <hip/hip_runtime.h>
#include <hip/hip_bf16.h>

// ---------------------------------------------------------------------------
// Product (vector) quantizer for MI455X / gfx1250.
//   z_e:       (B=8192, T=16, D=128) f32
//   codebooks: (T=16,  K=512, D=128) f32
// Outputs (flat, in order): z_q_st (B*T*D f32), tokens (B*T, as f32),
//                           vq_loss (1), utilization (1).
//
// Per-t nearest-code search is a GEMM (B x D) * (D x K) -> scores, then
// argmin_k( ||c_k||^2 - 2*z.c_k ).  GEMM runs on v_wmma_f32_16x16x32_bf16
// with a bf16 hi/lo split of both operands (hh + hl + lh) for ~fp32-accurate
// argmins.  The whole per-t codebook (256KB as bf16 hi/lo planes) is staged
// once per workgroup into CDNA5's 320KB LDS.
//
// Round-3 changes:
//  * amdgpu_waves_per_eu(2): the 259KB LDS footprint pins us at 8 waves/WGP
//    (2 per SIMD32); tell the register allocator so it stops serializing
//    ds_load bursts to save registers.
//  * Manual 2-deep software pipeline over N-tiles: tile j+1's 8 B-fragment
//    ds_load_b128 burst (and cnorm) issues before tile j's 12 WMMAs, so the
//    WMMA stream waits on loads a full compute block old instead of the
//    s_wait_dscnt 0x0 drain per WMMA pair seen in rounds 1-2.
// ---------------------------------------------------------------------------

typedef __attribute__((ext_vector_type(16))) __bf16 v16bf;
typedef __attribute__((ext_vector_type(8)))  float  v8f;

#define B_TOT   8192
#define T_TOT   16
#define K_TOT   512
#define D_TOT   128
#define ROWS_WG 256          // rows of z_e handled per workgroup (8 waves * 32)
#define BETA    0.25

// LDS layout (dynamic):
//   cb_hi  : K*D bf16   (128 KB)
//   cb_lo  : K*D bf16   (128 KB)
//   cnorm  : K   f32    (  2 KB)
//   wtok   : ROWS_WG int(  1 KB)
//   lred   : 8 f32
#define SMEM_BYTES ((size_t)K_TOT*D_TOT*2*2 + K_TOT*4 + ROWS_WG*4 + 8*4)

#define WMMA_BF16(A, Bm, C) \
    __builtin_amdgcn_wmma_f32_16x16x32_bf16(false, (A), false, (Bm), (short)0, (C), false, false)

__global__ void vq_init(unsigned int* __restrict__ used,
                        double* __restrict__ loss_acc) {
    int i = blockIdx.x * blockDim.x + threadIdx.x;
    if (i == 0) *loss_acc = 0.0;
    for (int j = i; j < T_TOT * K_TOT; j += blockDim.x * gridDim.x)
        used[j] = 0u;
}

// Load the 8 B fragments (hi/lo x 4 k-chunks) of one 16-column N-tile.
// Lane holds column n; each fragment is a contiguous 32B LDS read.
__device__ __forceinline__
void load_bfrags(const __bf16* cb_hi, const __bf16* cb_lo,
                 int n, int half, v16bf bh[4], v16bf bl[4])
{
    const v16bf* bhp = (const v16bf*)(cb_hi + (size_t)n * D_TOT + half * 16);
    const v16bf* blp = (const v16bf*)(cb_lo + (size_t)n * D_TOT + half * 16);
    #pragma unroll
    for (int c = 0; c < 4; ++c) {      // index 2c: chunk stride = 32 elems
        bh[c] = bhp[2 * c];
        bl[c] = blp[2 * c];
    }
}

// 12 WMMAs (hh/hl/lh chains) + running-argmin update for one N-tile.
__device__ __forceinline__
void score_update(const v16bf ahi[4], const v16bf alo[4],
                  const v16bf bh[4],  const v16bf bl[4],
                  float cn, int n, float minv[8], int minc[8])
{
    v8f acc0 = {}, acc1 = {}, acc2 = {};   // independent chains for ILP
    #pragma unroll
    for (int c = 0; c < 4; ++c) {
        acc0 = WMMA_BF16(ahi[c], bh[c], acc0);
        acc1 = WMMA_BF16(ahi[c], bl[c], acc1);
        acc2 = WMMA_BF16(alo[c], bh[c], acc2);
    }
    #pragma unroll
    for (int r = 0; r < 8; ++r) {
        float d = cn - 2.0f * (acc0[r] + acc1[r] + acc2[r]);
        if (d < minv[r]) { minv[r] = d; minc[r] = n; }   // first-hit kept
    }
}

__global__ __launch_bounds__(256)
__attribute__((amdgpu_waves_per_eu(2)))
void vq_main(const float* __restrict__ z_e,
             const float* __restrict__ cb,
             float* __restrict__ out_zq,
             float* __restrict__ out_tok,
             unsigned int* __restrict__ used,
             double* __restrict__ loss_acc)
{
    extern __shared__ char smem[];
    __bf16* cb_hi = (__bf16*)smem;                 // [K][D] hi plane
    __bf16* cb_lo = cb_hi + K_TOT * D_TOT;         // [K][D] lo plane
    float*  cnorm = (float*)(cb_lo + K_TOT * D_TOT);
    int*    wtok  = (int*)(cnorm + K_TOT);
    float*  lred  = (float*)(wtok + ROWS_WG);

    const int t       = blockIdx.x >> 5;           // 16 t values
    const int rb      = blockIdx.x & 31;           // 32 row blocks
    const int rowBase = rb * ROWS_WG;
    const int tid     = threadIdx.x;
    const int lane    = tid & 31;
    const int wave    = tid >> 5;
    const int half    = lane >> 4;                 // 0: lanes 0-15, 1: 16-31
    const int l16     = lane & 15;

    const float* cbt = cb + (size_t)t * K_TOT * D_TOT;

    // ---- Stage codebook into LDS as bf16 hi/lo planes + fp32 norms ----
    for (int i = tid; i < K_TOT * D_TOT; i += 256) {
        float f = cbt[i];
        __bf16 h = (__bf16)f;
        cb_hi[i] = h;
        cb_lo[i] = (__bf16)(f - (float)h);
    }
    for (int k = tid; k < K_TOT; k += 256) {
        const float4* cr = (const float4*)(cbt + (size_t)k * D_TOT);
        float s = 0.f;
        #pragma unroll 8
        for (int d = 0; d < D_TOT / 4; ++d) {
            float4 c4 = cr[d];
            s += c4.x*c4.x + c4.y*c4.y + c4.z*c4.z + c4.w*c4.w;
        }
        cnorm[k] = s;
    }
    __syncthreads();

    // ---- Pass 1: WMMA score sweep + running argmin ----
    // Wave handles 32 rows = 2 M-tiles of 16.
    const int waveRow = rowBase + wave * 32;

    for (int mt = 0; mt < 2; ++mt) {
        // A fragment row for this lane (ISA: lane L holds row M = L%16).
        const int rowA = waveRow + mt * 16 + l16;
        const float* zr = z_e + ((size_t)rowA * T_TOT + t) * D_TOT;

        // Load + split A tile: 4 k-chunks of 32, each a v16bf hi and lo.
        // Per-lane K order (16-bit A 16x32): half0 -> k {0..15}+kc,
        //                                    half1 -> k {16..31}+kc.
        v16bf ahi[4], alo[4];
        #pragma unroll
        for (int c = 0; c < 4; ++c) {
            const int off = c * 32 + half * 16;
            #pragma unroll
            for (int j = 0; j < 8; ++j) {
                float f0 = zr[off + j];
                float f1 = zr[off + 8 + j];
                __bf16 h0 = (__bf16)f0;
                __bf16 h1 = (__bf16)f1;
                ahi[c][j]     = h0;  alo[c][j]     = (__bf16)(f0 - (float)h0);
                ahi[c][8 + j] = h1;  alo[c][8 + j] = (__bf16)(f1 - (float)h1);
            }
        }

        // Running (min, argmin) in the WMMA C layout:
        //   VGPR slot r, this lane  ->  row M = r + 8*half, column = n.
        float minv[8];
        int   minc[8];
        #pragma unroll
        for (int r = 0; r < 8; ++r) { minv[r] = 3.4e38f; minc[r] = 0; }

        // 2-deep software pipeline over the 32 N-tiles.
        v16bf b0h[4], b0l[4], b1h[4], b1l[4];
        load_bfrags(cb_hi, cb_lo, l16, half, b0h, b0l);      // tile 0
        float cn0 = cnorm[l16];

        for (int nt = 0; nt < 32; nt += 2) {
            const int n0 = nt * 16 + l16;
            const int n1 = n0 + 16;
            // Prefetch tile nt+1 while tile nt computes.
            float cn1 = cnorm[n1];
            load_bfrags(cb_hi, cb_lo, n1, half, b1h, b1l);
            score_update(ahi, alo, b0h, b0l, cn0, n0, minv, minc);
            // Prefetch tile nt+2 (wraps to tile 0 on the last pass; the
            // redundant load keeps the body branch-free) while nt+1 computes.
            const int n2 = (nt + 2 < 32) ? (n1 + 16) : l16;
            cn0 = cnorm[n2];
            load_bfrags(cb_hi, cb_lo, n2, half, b0h, b0l);
            score_update(ahi, alo, b1h, b1l, cn1, n1, minv, minc);
        }

        // Butterfly argmin across the 16 lanes of each half (rows separate
        // per half).  Tie-break on lower code index = jnp.argmin semantics.
        #pragma unroll
        for (int m = 1; m < 16; m <<= 1) {
            #pragma unroll
            for (int r = 0; r < 8; ++r) {
                float ov = __shfl_xor(minv[r], m, 16);
                int   oc = __shfl_xor(minc[r], m, 16);
                if (ov < minv[r] || (ov == minv[r] && oc < minc[r])) {
                    minv[r] = ov; minc[r] = oc;
                }
            }
        }

        if (l16 == 0) {     // lane 0 (rows 0-7) and lane 16 (rows 8-15)
            #pragma unroll
            for (int r = 0; r < 8; ++r) {
                int lr   = wave * 32 + mt * 16 + half * 8 + r;
                int b    = rowBase + lr;
                int code = minc[r];
                wtok[lr] = code;
                out_tok[(size_t)b * T_TOT + t] = (float)code;
                used[t * K_TOT + code] = 1u;
            }
        }
    }
    __syncthreads();

    // ---- Pass 2: gather exact fp32 codes, emit z_q_st, accumulate loss ----
    // One wave = one row per iteration; lane covers 4 dims (coalesced float4).
    float lsum = 0.f;
    for (int i = 0; i < 32; ++i) {
        const int lr   = wave * 32 + i;
        const int b    = rowBase + lr;
        const int code = wtok[lr];
        const size_t zoff = ((size_t)b * T_TOT + t) * D_TOT;
        float4 z4 = ((const float4*)(z_e + zoff))[lane];
        float4 c4 = ((const float4*)(cbt + (size_t)code * D_TOT))[lane];
        float4 o;                                // z_e + (z_q - z_e), jnp order
        o.x = z4.x + (c4.x - z4.x);
        o.y = z4.y + (c4.y - z4.y);
        o.z = z4.z + (c4.z - z4.z);
        o.w = z4.w + (c4.w - z4.w);
        ((float4*)(out_zq + zoff))[lane] = o;
        float dx = z4.x - c4.x, dy = z4.y - c4.y,
              dz = z4.z - c4.z, dw = z4.w - c4.w;
        lsum += dx*dx + dy*dy + dz*dz + dw*dw;
    }
    #pragma unroll
    for (int m = 1; m < 32; m <<= 1) lsum += __shfl_xor(lsum, m, 32);
    if (lane == 0) lred[wave] = lsum;
    __syncthreads();
    if (tid == 0) {
        float s = 0.f;
        #pragma unroll
        for (int w = 0; w < 8; ++w) s += lred[w];
        atomicAdd(loss_acc, (double)s);          // global_atomic_add_f64
    }
}

__global__ void vq_finalize(const unsigned int* __restrict__ used,
                            const double* __restrict__ loss_acc,
                            float* __restrict__ out_loss,
                            float* __restrict__ out_util)
{
    __shared__ int sc[256];
    const int tid = threadIdx.x;
    int c = 0;
    for (int i = tid; i < T_TOT * K_TOT; i += 256) c += (used[i] != 0u);
    sc[tid] = c;
    __syncthreads();
    for (int s = 128; s > 0; s >>= 1) {
        if (tid < s) sc[tid] += sc[tid + s];
        __syncthreads();
    }
    if (tid == 0) {
        *out_loss = (float)(BETA * (*loss_acc) /
                            ((double)B_TOT * T_TOT * D_TOT));
        *out_util = (float)sc[0] / (float)(T_TOT * K_TOT);
    }
}

extern "C" void kernel_launch(void* const* d_in, const int* in_sizes, int n_in,
                              void* d_out, int out_size, void* d_ws, size_t ws_size,
                              hipStream_t stream)
{
    const float* z_e = (const float*)d_in[0];
    const float* cb  = (const float*)d_in[1];

    float* out       = (float*)d_out;
    float* out_zq    = out;                                       // B*T*D
    float* out_tok   = out + (size_t)B_TOT * T_TOT * D_TOT;       // B*T
    float* out_loss  = out_tok + (size_t)B_TOT * T_TOT;           // 1
    float* out_util  = out_loss + 1;                              // 1

    unsigned int* used = (unsigned int*)d_ws;                     // T*K flags
    double* loss_acc   = (double*)((char*)d_ws +
                                   (size_t)T_TOT * K_TOT * sizeof(unsigned int));

    vq_init<<<1, 256, 0, stream>>>(used, loss_acc);
    vq_main<<<dim3(T_TOT * (B_TOT / ROWS_WG)), 256, SMEM_BYTES, stream>>>(
        z_e, cb, out_zq, out_tok, used, loss_acc);
    vq_finalize<<<1, 256, 0, stream>>>(used, loss_acc, out_loss, out_util);
}
